// Attention_57672820850902
// MI455X (gfx1250) — matrix operand order repaired and checked by reading, not compiled, not run on hardware
//
#include <hip/hip_runtime.h>
#include <hip/hip_bf16.h>

typedef __bf16 bf16;
typedef __attribute__((ext_vector_type(16))) __bf16 v16bf;
typedef __attribute__((ext_vector_type(8)))  __bf16 v8bf;
typedef __attribute__((ext_vector_type(8)))  float  v8f;

#define HW    16384
#define IMW   128
#define CCH   192
#define NHEAD 6
#define NB    4

static __device__ __forceinline__ v8f wmma_bf16(v16bf a, v16bf b, v8f c) {
  // D = A(16x32 bf16) * B(32x16 bf16) + C(16x16 f32)
  return __builtin_amdgcn_wmma_f32_16x16x32_bf16(false, a, false, b, (short)0, c,
                                                 false, false);
}

// A fragment from k-contiguous bf16 row: lane half picks K block
// {2*ks..2*ks+15}, ks = 8*(lane>=16)  (contiguous k order, matches B)
static __device__ __forceinline__ v16bf load_a16(const bf16* wr, int ks) {
  v16bf a;
#pragma unroll
  for (int i = 0; i < 8; ++i) { a[i] = wr[2 * ks + i]; a[i + 8] = wr[2 * ks + 8 + i]; }
  return a;
}

// B fragment: 16 k-contiguous bf16 from LDS, as two aligned 16B chunks.
static __device__ __forceinline__ v16bf lds_load16(const bf16* p) {
  v8bf lo = *(const v8bf*)p;
  v8bf hi = *(const v8bf*)(p + 8);
  v16bf r;
#pragma unroll
  for (int i = 0; i < 8; ++i) { r[i] = lo[i]; r[i + 8] = hi[i]; }
  return r;
}

// ---------------------------------------------------------------------------
// Weight prep: bf16-pack kv_w/q_w/proj_w ([M][K], already k-contiguous) and
// repack q_dw_w [Cout][Cin][3][3] -> [tap][Cout][Cin] bf16 (k-contiguous A).
// ---------------------------------------------------------------------------
__global__ void pack_weights_kernel(const float* __restrict__ kv_w,
                                    const float* __restrict__ q_w,
                                    const float* __restrict__ proj_w,
                                    const float* __restrict__ q_dw_w,
                                    bf16* __restrict__ wkv, bf16* __restrict__ wq,
                                    bf16* __restrict__ wproj,
                                    bf16* __restrict__ wq3) {
  const int i = blockIdx.x * 256 + threadIdx.x;
  if (i < 384 * CCH) wkv[i] = (bf16)kv_w[i];
  if (i < CCH * CCH) { wq[i] = (bf16)q_w[i]; wproj[i] = (bf16)proj_w[i]; }
  if (i < CCH * CCH * 9) {
    const int tap = i % 9;
    const int cin = (i / 9) % CCH;
    const int m   = i / (9 * CCH);
    wq3[((size_t)tap * CCH + m) * CCH + cin] = (bf16)q_dw_w[i];
  }
}

// ---------------------------------------------------------------------------
// Cooperative 1x1 conv: OUT_bf16[b][m][n] = sum_k W[m][k] X_f32[b][k][n]
// block = 256 thr (8 waves). Block tile: 96 M rows (waves 0..5) x 64 N.
// B tile staged transposed into LDS (k-contiguous, padded rows) per K-step.
// grid (HW/64, M/96, B)
// ---------------------------------------------------------------------------
__global__ void conv1x1_kernel(const float* __restrict__ X,
                               const bf16* __restrict__ Wb,
                               bf16* __restrict__ OUT, int M, int K) {
  __shared__ bf16 sB[64][40];  // [n][k], rows padded to 40 (20-dword stride)
  const int tid  = threadIdx.x;
  const int wave = tid >> 5, lane = tid & 31;
  const int l16  = lane & 15, half = lane >> 4, ks = half * 8;
  const int n0   = blockIdx.x * 64;
  const int m0   = blockIdx.y * 96 + wave * 16;
  const int b    = blockIdx.z;

  v8f acc[4] = {};
  for (int k0 = 0; k0 < K; k0 += 32) {
    __syncthreads();
    {  // stage 32k x 64n fp32 -> bf16 transposed; coalesced float4 reads
      const int kk = tid >> 3;
      const int nb = (tid & 7) * 8;
      const float* src = X + ((size_t)b * K + k0 + kk) * HW + n0 + nb;
      const float4 f0 = *(const float4*)src;
      const float4 f1 = *(const float4*)(src + 4);
      sB[nb + 0][kk] = (bf16)f0.x; sB[nb + 1][kk] = (bf16)f0.y;
      sB[nb + 2][kk] = (bf16)f0.z; sB[nb + 3][kk] = (bf16)f0.w;
      sB[nb + 4][kk] = (bf16)f1.x; sB[nb + 5][kk] = (bf16)f1.y;
      sB[nb + 6][kk] = (bf16)f1.z; sB[nb + 7][kk] = (bf16)f1.w;
    }
    __syncthreads();
    if (wave < 6) {
      const v16bf a = load_a16(Wb + (size_t)(m0 + l16) * K + k0, ks);
#pragma unroll
      for (int j = 0; j < 4; ++j) {
        const v16bf bb = lds_load16(&sB[j * 16 + l16][half * 16]);
        acc[j] = wmma_bf16(a, bb, acc[j]);
      }
    }
  }
  if (wave < 6) {
#pragma unroll
    for (int j = 0; j < 4; ++j) {
      bf16* o = OUT + ((size_t)b * M + m0 + half * 8) * HW + n0 + j * 16 + l16;
#pragma unroll
      for (int i = 0; i < 8; ++i) o[(size_t)i * HW] = (bf16)acc[j][i];
    }
  }
}

// ---------------------------------------------------------------------------
// Depthwise 3x3, SAME padding: bf16 in/out, fp32 weights [Ctot][1][3][3]
// ---------------------------------------------------------------------------
__global__ void dw3x3_kernel(const bf16* __restrict__ IN,
                             const float* __restrict__ Wdw,
                             bf16* __restrict__ OUT, int Ctot) {
  size_t idx = (size_t)blockIdx.x * blockDim.x + threadIdx.x;
  size_t total = (size_t)NB * Ctot * HW;
  if (idx >= total) return;
  const int n = (int)(idx & (HW - 1));
  const int c = (int)((idx >> 14) % (unsigned)Ctot);
  const int h = n >> 7, w = n & 127;
  const bf16* src = IN + (idx - n);
  const float* kw = Wdw + (size_t)c * 9;
  float s = 0.f;
#pragma unroll
  for (int dy = 0; dy < 3; ++dy) {
    const int hh = h + dy - 1;
    if ((unsigned)hh >= 128u) continue;
#pragma unroll
    for (int dx = 0; dx < 3; ++dx) {
      const int ww = w + dx - 1;
      if ((unsigned)ww >= 128u) continue;
      s += kw[dy * 3 + dx] * (float)src[hh * IMW + ww];
    }
  }
  OUT[idx] = (bf16)s;
}

// ---------------------------------------------------------------------------
// Cooperative full 3x3 conv 192->192, implicit GEMM K=192*9.
// Block tile 96 M x 64 N; per K-step stage a 3-row x 66-pixel x 32-ch halo
// tile (zero-filled at borders) reused by 9 taps x 6 m-tiles.
// Wb layout [tap][Cout][Cin] bf16.  grid (HW/64, 2, B), block 256.
// ---------------------------------------------------------------------------
__global__ void conv3x3_kernel(const bf16* __restrict__ X,
                               const bf16* __restrict__ Wb,
                               bf16* __restrict__ OUT) {
  __shared__ bf16 sX[3][66][40];  // [row][halo w][k], padded k rows
  const int tid  = threadIdx.x;
  const int wave = tid >> 5, lane = tid & 31;
  const int l16  = lane & 15, half = lane >> 4, ks = half * 8;
  const int n0   = blockIdx.x * 64;
  const int h    = n0 >> 7;        // 64-pixel tile lies inside one image row
  const int w0   = n0 & 127;
  const int m0   = blockIdx.y * 96 + wave * 16;
  const int b    = blockIdx.z;

  v8f acc[4] = {};
  for (int k0 = 0; k0 < CCH; k0 += 32) {
    __syncthreads();
    // stage: e -> (w fastest for coalesced global reads), zero-fill OOB
    for (int e = tid; e < 3 * 66 * 32; e += 256) {
      const int w  = e % 66;
      const int kr = e / 66;
      const int kk = kr & 31;
      const int r  = kr >> 5;
      const int hh = h + r - 1;
      const int ww = w0 - 1 + w;
      bf16 v = (bf16)0.0f;
      if ((unsigned)hh < 128u && (unsigned)ww < 128u)
        v = X[((size_t)b * CCH + k0 + kk) * HW + hh * IMW + ww];
      sX[r][w][kk] = v;
    }
    __syncthreads();
    if (wave < 6) {
      for (int tap = 0; tap < 9; ++tap) {
        const int dy = tap / 3, dx = tap % 3;  // staged row r=dy, w shift dx
        const v16bf a =
            load_a16(Wb + ((size_t)tap * CCH + m0 + l16) * CCH + k0, ks);
#pragma unroll
        for (int j = 0; j < 4; ++j) {
          const v16bf bb = lds_load16(&sX[dy][j * 16 + l16 + dx][half * 16]);
          acc[j] = wmma_bf16(a, bb, acc[j]);
        }
      }
    }
  }
  if (wave < 6) {
#pragma unroll
    for (int j = 0; j < 4; ++j) {
      bf16* o = OUT + ((size_t)b * CCH + m0 + half * 8) * HW + n0 + j * 16 + l16;
#pragma unroll
      for (int i = 0; i < 8; ++i) o[(size_t)i * HW] = (bf16)acc[j][i];
    }
  }
}

// ---------------------------------------------------------------------------
// Row L2 norms: rows 0..767 = q channels, 768..1535 = k channels.
// rnorm[r] = 1 / max(||row||, 1e-12)
// ---------------------------------------------------------------------------
__global__ void norm_kernel(const bf16* __restrict__ Q,
                            const bf16* __restrict__ KV,
                            float* __restrict__ rnorm) {
  const int r = blockIdx.x;
  const bf16* src;
  if (r < NB * CCH) {
    src = Q + (size_t)r * HW;
  } else {
    const int rr = r - NB * CCH;
    const int b = rr / CCH, c = rr % CCH;
    src = KV + ((size_t)b * 2 * CCH + c) * HW;
  }
  float s = 0.f;
  for (int i = threadIdx.x; i < HW; i += 256) {
    const float v = (float)src[i];
    s += v * v;
  }
  __shared__ float red[256];
  red[threadIdx.x] = s;
  __syncthreads();
  for (int off = 128; off > 0; off >>= 1) {
    if (threadIdx.x < off) red[threadIdx.x] += red[threadIdx.x + off];
    __syncthreads();
  }
  if (threadIdx.x == 0) rnorm[r] = 1.f / fmaxf(sqrtf(red[0]), 1e-12f);
}

// ---------------------------------------------------------------------------
// attn = softmax((q_hat k_hat^T) * temp): K=16384 split over 8 waves,
// LDS reduce + softmax. Fragments are contiguous 32B loads. grid 24, blk 256.
// ---------------------------------------------------------------------------
__global__ void attn_kernel(const bf16* __restrict__ Q,
                            const bf16* __restrict__ KV,
                            const float* __restrict__ rnorm,
                            const float* __restrict__ temp,
                            float* __restrict__ ATTN) {
  const int bh = blockIdx.x;
  const int b = bh / NHEAD, h = bh % NHEAD;
  const int wave = threadIdx.x >> 5;
  const int lane = threadIdx.x & 31;
  const int l16 = lane & 15, half = lane >> 4, ks = half * 8;

  const bf16* qb = Q  + ((size_t)b * CCH + h * 32) * HW;
  const bf16* kb = KV + ((size_t)b * 2 * CCH + h * 32) * HW;

  v8f acc[2][2] = {};
  for (int t = wave; t < HW / 32; t += 8) {
    const int n0 = t * 32;
    v16bf af[2], bfm[2];
#pragma unroll
    for (int mi = 0; mi < 2; ++mi) {
      const bf16* p = qb + (size_t)(mi * 16 + l16) * HW + n0 + 2 * ks;
#pragma unroll
      for (int i = 0; i < 8; ++i) { af[mi][i] = p[i]; af[mi][i + 8] = p[8 + i]; }
    }
#pragma unroll
    for (int ni = 0; ni < 2; ++ni)
      bfm[ni] = *(const v16bf*)(kb + (size_t)(ni * 16 + l16) * HW + n0 + half * 16);
#pragma unroll
    for (int mi = 0; mi < 2; ++mi)
#pragma unroll
      for (int ni = 0; ni < 2; ++ni)
        acc[mi][ni] = wmma_bf16(af[mi], bfm[ni], acc[mi][ni]);
  }

  __shared__ float part[8][32][32];
#pragma unroll
  for (int mi = 0; mi < 2; ++mi)
#pragma unroll
    for (int ni = 0; ni < 2; ++ni)
#pragma unroll
      for (int i = 0; i < 8; ++i)
        part[wave][mi * 16 + half * 8 + i][ni * 16 + l16] = acc[mi][ni][i];
  __syncthreads();

  __shared__ float smat[32][32];
  const int rbase = b * CCH + h * 32;
  for (int e = threadIdx.x; e < 1024; e += 256) {
    const int c = e >> 5, d = e & 31;
    float s = 0.f;
#pragma unroll
    for (int w = 0; w < 8; ++w) s += part[w][c][d];
    smat[c][d] = s * rnorm[rbase + c] * rnorm[NB * CCH + rbase + d] * temp[h];
  }
  __syncthreads();
  if (threadIdx.x < 32) {
    const int c = threadIdx.x;
    float mx = -3.4e38f;
    for (int d = 0; d < 32; ++d) mx = fmaxf(mx, smat[c][d]);
    float ex[32], sum = 0.f;
    for (int d = 0; d < 32; ++d) { ex[d] = __expf(smat[c][d] - mx); sum += ex[d]; }
    const float inv = 1.f / sum;
    float* o = ATTN + ((size_t)bh * 32 + c) * 32;
    for (int d = 0; d < 32; ++d) o[d] = ex[d] * inv;
  }
}

// ---------------------------------------------------------------------------
// Z[b][h*32+c][n] = sum_d attn[b][h][c][d] * v[b][h*32+d][n]  (single K step)
// grid (HW/64, 24), block 32
// ---------------------------------------------------------------------------
__global__ void av_kernel(const float* __restrict__ ATTN,
                          const bf16* __restrict__ KV,
                          bf16* __restrict__ Z) {
  const int bh = blockIdx.y;
  const int b = bh / NHEAD, h = bh % NHEAD;
  const int lane = threadIdx.x;
  const int l16 = lane & 15, half = lane >> 4, ks = half * 8;
  const int n0 = blockIdx.x * 64;

  const float* am = ATTN + (size_t)bh * 1024;
  v16bf af[2];
#pragma unroll
  for (int mi = 0; mi < 2; ++mi) {
    const float* row = am + (size_t)(mi * 16 + l16) * 32;
#pragma unroll
    for (int i = 0; i < 8; ++i) {
      af[mi][i]     = (bf16)row[2 * ks + i];
      af[mi][i + 8] = (bf16)row[2 * ks + 8 + i];
    }
  }
  const bf16* vb = KV + ((size_t)b * 2 * CCH + CCH + h * 32) * HW;
  v8f acc[2][4] = {};
#pragma unroll
  for (int ni = 0; ni < 4; ++ni) {
    const int n = n0 + ni * 16 + l16;
    v16bf bb;
#pragma unroll
    for (int i = 0; i < 16; ++i) bb[i] = vb[(size_t)(half * 16 + i) * HW + n];
#pragma unroll
    for (int mi = 0; mi < 2; ++mi) acc[mi][ni] = wmma_bf16(af[mi], bb, acc[mi][ni]);
  }
  bf16* zb = Z + ((size_t)b * CCH + h * 32) * HW;
#pragma unroll
  for (int mi = 0; mi < 2; ++mi)
#pragma unroll
    for (int ni = 0; ni < 4; ++ni)
#pragma unroll
      for (int i = 0; i < 8; ++i)
        zb[(size_t)(mi * 16 + half * 8 + i) * HW + n0 + ni * 16 + l16] =
            (bf16)acc[mi][ni][i];
}

// ---------------------------------------------------------------------------
// Cooperative proj + residual: OUT_f32 = x - proj_w @ Z (Z bf16).
// Same block structure as conv1x1. grid (HW/64, 2, B), block 256.
// ---------------------------------------------------------------------------
__global__ void proj_kernel(const bf16* __restrict__ Z,
                            const bf16* __restrict__ Wb,
                            const float* __restrict__ Xres,
                            float* __restrict__ OUT) {
  __shared__ bf16 sB[64][40];
  const int tid  = threadIdx.x;
  const int wave = tid >> 5, lane = tid & 31;
  const int l16  = lane & 15, half = lane >> 4, ks = half * 8;
  const int n0   = blockIdx.x * 64;
  const int m0   = blockIdx.y * 96 + wave * 16;
  const int b    = blockIdx.z;

  v8f acc[4] = {};
  for (int k0 = 0; k0 < CCH; k0 += 32) {
    __syncthreads();
    {  // stage bf16 tile transposed; coalesced 16B reads
      const int kk = tid >> 3;
      const int nb = (tid & 7) * 8;
      const v8bf v = *(const v8bf*)(Z + ((size_t)b * CCH + k0 + kk) * HW + n0 + nb);
#pragma unroll
      for (int i = 0; i < 8; ++i) sB[nb + i][kk] = v[i];
    }
    __syncthreads();
    if (wave < 6) {
      const v16bf a = load_a16(Wb + (size_t)(m0 + l16) * CCH + k0, ks);
#pragma unroll
      for (int j = 0; j < 4; ++j) {
        const v16bf bb = lds_load16(&sB[j * 16 + l16][half * 16]);
        acc[j] = wmma_bf16(a, bb, acc[j]);
      }
    }
  }
  if (wave < 6) {
#pragma unroll
    for (int j = 0; j < 4; ++j) {
      const size_t base =
          ((size_t)b * CCH + m0 + half * 8) * HW + n0 + j * 16 + l16;
#pragma unroll
      for (int i = 0; i < 8; ++i)
        OUT[base + (size_t)i * HW] = Xres[base + (size_t)i * HW] - acc[j][i];
    }
  }
}

// ---------------------------------------------------------------------------
extern "C" void kernel_launch(void* const* d_in, const int* in_sizes, int n_in,
                              void* d_out, int out_size, void* d_ws,
                              size_t ws_size, hipStream_t stream) {
  const float* x        = (const float*)d_in[0];
  const float* y        = (const float*)d_in[1];
  const float* kv_w     = (const float*)d_in[2];  // [384,192,1,1]
  const float* kv_dw_w  = (const float*)d_in[3];  // [384,1,3,3]
  const float* q_w      = (const float*)d_in[4];  // [192,192,1,1]
  const float* q_dw_w   = (const float*)d_in[5];  // [192,192,3,3]
  const float* proj_w   = (const float*)d_in[6];  // [192,192,1,1]
  const float* temperature = (const float*)d_in[7];  // [6]
  float* out = (float*)d_out;

  // Workspace layout (bytes), regions reused across the serialized stream:
  //  [0,         50331648)  kv_pre bf16 -> later q_pre(25MB)+q(25MB) -> Z(25MB)
  //  [50331648, 100663296)  kv bf16 (k = first 192 ch, v = next 192 ch)
  //  [100663296,100669440)  rnorm fp32 [1536]
  //  [100671488,100769792)  attn fp32 [4*6*32*32]
  //  [100769792,101728256)  packed bf16 weights (kv | q | proj | q3x3 repack)
  char* ws = (char*)d_ws;
  bf16*  kvpre = (bf16*)ws;
  bf16*  kv    = (bf16*)(ws + 50331648);
  bf16*  qpre  = (bf16*)ws;
  bf16*  q     = (bf16*)(ws + 25165824);
  float* rnorm = (float*)(ws + 100663296);
  float* attn  = (float*)(ws + 100671488);
  bf16*  z     = (bf16*)ws;
  bf16*  wkv   = (bf16*)(ws + 100769792);
  bf16*  wq    = (bf16*)(ws + 100917248);
  bf16*  wproj = (bf16*)(ws + 100990976);
  bf16*  wq3   = (bf16*)(ws + 101064704);

  // 0) pack weights to bf16 (and repack 3x3 to [tap][m][cin])
  pack_weights_kernel<<<(CCH * CCH * 9 + 255) / 256, 256, 0, stream>>>(
      kv_w, q_w, proj_w, q_dw_w, wkv, wq, wproj, wq3);
  // 1) kv_pre = conv1x1(x, kv_w)   (M=384)
  conv1x1_kernel<<<dim3(HW / 64, 384 / 96, NB), 256, 0, stream>>>(x, wkv, kvpre,
                                                                  384, CCH);
  // 2) kv = depthwise3x3(kv_pre)
  {
    const size_t tot = (size_t)NB * 384 * HW;
    dw3x3_kernel<<<(unsigned)((tot + 255) / 256), 256, 0, stream>>>(
        kvpre, kv_dw_w, kv, 384);
  }
  // 3) q_pre = conv1x1(y, q_w)   (kv_pre region dead -> reuse)
  conv1x1_kernel<<<dim3(HW / 64, CCH / 96, NB), 256, 0, stream>>>(y, wq, qpre,
                                                                  CCH, CCH);
  // 4) q = full conv3x3(q_pre, q_dw_w)
  conv3x3_kernel<<<dim3(HW / 64, CCH / 96, NB), 256, 0, stream>>>(qpre, wq3, q);
  // 5) row norms of q and k
  norm_kernel<<<2 * NB * CCH, 256, 0, stream>>>(q, kv, rnorm);
  // 6) attn = softmax(q_hat k_hat^T * temp)
  attn_kernel<<<NB * NHEAD, 256, 0, stream>>>(q, kv, rnorm, temperature, attn);
  // 7) z = attn @ v   (q_pre region dead -> reuse for z)
  av_kernel<<<dim3(HW / 64, NB * NHEAD), 32, 0, stream>>>(attn, kv, z);
  // 8) out = x - proj_w @ z
  proj_kernel<<<dim3(HW / 64, CCH / 96, NB), 256, 0, stream>>>(z, wproj, x, out);
}
